// GQA_88476326298085
// MI455X (gfx1250) — compile-verified, matrix operand order checked by reading, not run on hardware
//
#include <hip/hip_runtime.h>
#include <hip/hip_bf16.h>
#include <stdint.h>

typedef __bf16 bf16_t;
typedef __attribute__((ext_vector_type(16))) __bf16        v16bf;
typedef __attribute__((ext_vector_type(8)))  float         v8f;
typedef __attribute__((ext_vector_type(4)))  unsigned int  v4u;

union BFrag { v16bf v; v4u u[2]; };

#define DM     2048
#define NH     16
#define NKV    4
#define HD     128
#define NREP   4
#define BB     2
#define TT     2048
#define HALF_D 64

// --------------------------- CDNA5 async copy path -------------------------
#if defined(__has_builtin)
#if __has_builtin(__builtin_amdgcn_global_load_async_to_lds_b128) && \
    __has_builtin(__builtin_amdgcn_s_wait_asynccnt)
#define HAVE_ASYNC 1
#endif
#endif
#ifndef HAVE_ASYNC
#define HAVE_ASYNC 0
#endif

// Builtin signature (probed): (int4 AS(1)* gsrc, int4 AS(3)* lds, imm, imm cpol)
typedef int v4i_t __attribute__((vector_size(16)));
typedef __attribute__((address_space(1))) v4i_t as1_v4i;
typedef __attribute__((address_space(3))) v4i_t as3_v4i;

__device__ __forceinline__ void async_copy16(bf16_t* lds, const bf16_t* g) {
#if HAVE_ASYNC
  __builtin_amdgcn_global_load_async_to_lds_b128((as1_v4i*)g, (as3_v4i*)lds, 0, 0);
#else
  *(v4u*)lds = *(const v4u*)g;   // fallback: global_load_b128 + ds_store_b128
#endif
}

__device__ __forceinline__ void async_join_barrier() {
#if HAVE_ASYNC
  __builtin_amdgcn_s_wait_asynccnt(0);
#endif
  __syncthreads();
}

// ---------------------------------------------------------------------------
// elementwise f32 -> bf16 cast (grid stride)
// ---------------------------------------------------------------------------
__global__ __launch_bounds__(256)
void cast_bf16_kernel(const float* __restrict__ in, bf16_t* __restrict__ out, size_t n) {
  size_t i = (size_t)blockIdx.x * blockDim.x + threadIdx.x;
  const size_t stride = (size_t)gridDim.x * blockDim.x;
  for (; i < n; i += stride) out[i] = (bf16_t)in[i];
}

// ---------------------------------------------------------------------------
// W (K x N) f32 row-major  ->  Wt (N x K) bf16 row-major, via LDS 32x32 tile
// ---------------------------------------------------------------------------
__global__ __launch_bounds__(256)
void transpose_cast_kernel(const float* __restrict__ W, bf16_t* __restrict__ Wt,
                           int K, int N) {
  __shared__ float tile[32][33];
  const int bk = blockIdx.x * 32, bn = blockIdx.y * 32;
  const int tx = threadIdx.x & 31, ty = threadIdx.x >> 5;
  #pragma unroll
  for (int r = ty; r < 32; r += 8)
    tile[r][tx] = W[(size_t)(bk + r) * N + bn + tx];
  __syncthreads();
  #pragma unroll
  for (int r = ty; r < 32; r += 8)
    Wt[(size_t)(bn + r) * K + bk + tx] = (bf16_t)tile[tx][r];
}

// ---------------------------------------------------------------------------
// C (M x N) f32 = A (M x K) bf16 @ Bt (N x K) bf16   [Bt = B transposed]
// Block: 256 threads (8 waves), tile 256M x 64N x 32K, async double-buffered
// LDS staging. Each wave: 32 rows x 64 cols = 2 A-frags x 4 B-frags = 8 WMMA
// per k-step from 12 ds_load_b128 (all fragments live -> partial DScnt waits).
// ---------------------------------------------------------------------------
__global__ __launch_bounds__(256)
void gemm_bf16_wmma(const bf16_t* __restrict__ A, const bf16_t* __restrict__ Bt,
                    float* __restrict__ C, int M, int N, int K) {
  __shared__ __align__(16) bf16_t shA[2][256 * 32];   // 2 x 16 KB
  __shared__ __align__(16) bf16_t shB[2][64 * 32];    // 2 x  4 KB
  const int tid  = threadIdx.x;
  const int wid  = tid >> 5;
  const int lane = tid & 31;
  const int half = lane >> 4, idx = lane & 15;
  const int nbn  = N >> 6;
  const int bm0  = (int)(blockIdx.x / nbn) * 256;
  const int bn0  = (int)(blockIdx.x % nbn) * 64;

  // stage one 32-wide k-chunk into LDS buffer `buf`
  auto stage = [&](int kk, int buf) {
    #pragma unroll
    for (int s = 0; s < 4; ++s) {               // A: 1024 x 16B segments
      const int seg = tid + s * 256;
      const int r = seg >> 2, cs = (seg & 3) * 8;
      async_copy16(&shA[buf][r * 32 + cs], A + (size_t)(bm0 + r) * K + kk + cs);
    }
    {                                            // B: 256 x 16B segments
      const int r = tid >> 2, cs = (tid & 3) * 8;
      async_copy16(&shB[buf][r * 32 + cs], Bt + (size_t)(bn0 + r) * K + kk + cs);
    }
  };

  stage(0, 0);
  v8f acc[2][4] = {};
  int buf = 0;
  for (int kk = 0; kk < K; kk += 32) {
    async_join_barrier();                       // buffer `buf` ready for all waves
    if (kk + 32 < K) stage(kk + 32, buf ^ 1);   // prefetch next chunk (overlaps WMMA)

    BFrag a[2], b[4];
    #pragma unroll
    for (int i = 0; i < 2; ++i) {
      const bf16_t* arow = &shA[buf][(wid * 32 + i * 16 + idx) * 32];
      a[i].u[0] = *(const v4u*)(arow + 8 * half);
      a[i].u[1] = *(const v4u*)(arow + 16 + 8 * half);
    }
    #pragma unroll
    for (int j = 0; j < 4; ++j) {
      const bf16_t* brow = &shB[buf][(j * 16 + idx) * 32];
      b[j].u[0] = *(const v4u*)(brow + 8 * half);
      b[j].u[1] = *(const v4u*)(brow + 16 + 8 * half);
    }
    #pragma unroll
    for (int i = 0; i < 2; ++i)
      #pragma unroll
      for (int j = 0; j < 4; ++j)
        acc[i][j] = __builtin_amdgcn_wmma_f32_16x16x32_bf16(
            false, a[i].v, false, b[j].v, (short)0, acc[i][j], false, false);
    buf ^= 1;
  }

  #pragma unroll
  for (int i = 0; i < 2; ++i)
    #pragma unroll
    for (int j = 0; j < 4; ++j) {
      const int col = bn0 + j * 16 + idx;
      #pragma unroll
      for (int r = 0; r < 8; ++r) {
        const int m = bm0 + wid * 32 + i * 16 + r + 8 * half;
        C[(size_t)m * N + col] = acc[i][j][r];
      }
    }
}

// ---------------------------------------------------------------------------
// RoPE + cast + relayout: proj (B*T, nheads*128) f32 -> out (b, h, t, d) bf16
// ---------------------------------------------------------------------------
__global__ __launch_bounds__(256)
void rope_qk_kernel(const float* __restrict__ proj, const float* __restrict__ cosb,
                    const float* __restrict__ sinb, bf16_t* __restrict__ out,
                    int nheads) {
  size_t i = (size_t)blockIdx.x * blockDim.x + threadIdx.x;
  const size_t total = (size_t)BB * TT * nheads * HALF_D;
  if (i >= total) return;
  const int d = (int)(i & (HALF_D - 1)); i >>= 6;
  const int h = (int)(i % nheads);       i /= nheads;
  const int t = (int)(i % TT);
  const int b = (int)(i / TT);
  const float* src = proj + ((size_t)(b * TT + t) * nheads + h) * HD;
  const float x1 = src[d], x2 = src[d + HALF_D];
  const float c = cosb[t * HALF_D + d], s = sinb[t * HALF_D + d];
  bf16_t* dst = out + (((size_t)(b * nheads + h) * TT) + t) * HD;
  dst[d]          = (bf16_t)(x1 * c - x2 * s);
  dst[d + HALF_D] = (bf16_t)(x2 * c + x1 * s);
}

// ---------------------------------------------------------------------------
// V (b,t,g,d) f32 -> Vt (b,g,d,t) bf16 via LDS 32x32 transpose
// grid: (T/32, HD/32, B*NKV)
// ---------------------------------------------------------------------------
__global__ __launch_bounds__(256)
void vtrans_kernel(const float* __restrict__ vf, bf16_t* __restrict__ vt) {
  __shared__ float tile[32][33];
  const int bg = blockIdx.z;
  const int b = bg >> 2, g = bg & 3;
  const int t0 = blockIdx.x * 32, d0 = blockIdx.y * 32;
  const int tx = threadIdx.x & 31, ty = threadIdx.x >> 5;
  #pragma unroll
  for (int r = ty; r < 32; r += 8)
    tile[r][tx] = vf[(size_t)(b * TT + t0 + r) * (NKV * HD) + g * HD + d0 + tx];
  __syncthreads();
  #pragma unroll
  for (int r = ty; r < 32; r += 8)
    vt[((size_t)bg * HD + d0 + r) * TT + t0 + tx] = (bf16_t)tile[tx][r];
}

// ---------------------------------------------------------------------------
// Flash attention, block-cooperative: one block = (b, h, 128 queries).
// 8 waves x 16-query tiles share the (b,g) KV stream: each 32-key K-tile
// (32x128) and V-tile (128x32, d-major) is async-staged once into
// double-buffered LDS and consumed by all 8 waves. Waves past their causal
// limit skip compute (wave-uniform branch) but join staging + barriers.
// ---------------------------------------------------------------------------
__global__ __launch_bounds__(256)
void flash_attn_kernel(const bf16_t* __restrict__ qbf,
                       const bf16_t* __restrict__ kbf,
                       const bf16_t* __restrict__ vtbf,
                       bf16_t* __restrict__ attnbf) {
  __shared__ __align__(16) bf16_t shK[2][32 * HD];    // 2 x 8 KB, [key][d]
  __shared__ __align__(16) bf16_t shV[2][HD * 32];    // 2 x 8 KB, [d][key]
  __shared__ __align__(16) bf16_t pshare[8][16 * 32]; // 8 KB
  const int tid  = threadIdx.x;
  const int wid  = tid >> 5;
  const int lane = tid & 31;
  const int half = lane >> 4;
  const int idx  = lane & 15;

  const int nqb  = TT / 128;                 // query blocks per (b,h)
  const int qblk = (int)(blockIdx.x % nqb);
  const int bh   = (int)(blockIdx.x / nqb);
  const int h = bh % NH;
  const int b = bh / NH;
  const int g = h / NREP;
  const int qt = qblk * 8 + wid;             // this wave's 16-row query tile

  const float sc = 0.088388347648318447f * 1.4426950408889634f; // 1/sqrt(128)*log2(e)

  const bf16_t* kbase = kbf  + ((size_t)(b * NKV + g) * TT) * HD;
  const bf16_t* vbase = vtbf + ((size_t)(b * NKV + g) * HD) * TT;

  // stage K-tile + V-tile for key block kt into buffer `buf` (1024 segs / 256 thr)
  auto stage = [&](int kt, int buf) {
    #pragma unroll
    for (int s = 0; s < 2; ++s) {            // K: 32 rows x 16 segs
      const int seg = tid + s * 256;
      const int r = seg >> 4, cs = (seg & 15) * 8;
      async_copy16(&shK[buf][r * HD + cs],
                   kbase + (size_t)(kt * 32 + r) * HD + cs);
    }
    #pragma unroll
    for (int s = 0; s < 2; ++s) {            // V: 128 rows (d) x 4 segs
      const int seg = tid + s * 256;
      const int d = seg >> 2, cs = (seg & 3) * 8;
      async_copy16(&shV[buf][d * 32 + cs],
                   vbase + (size_t)d * TT + kt * 32 + cs);
    }
  };

  // Q fragments (A layout), row = qt*16 + idx
  const bf16_t* qrow = qbf + (((size_t)bh * TT) + (size_t)qt * 16 + idx) * HD;
  BFrag qf[4];
  #pragma unroll
  for (int c = 0; c < 4; ++c) {
    qf[c].u[0] = *(const v4u*)(qrow + c * 32 + 8 * half);
    qf[c].u[1] = *(const v4u*)(qrow + c * 32 + 16 + 8 * half);
  }

  v8f o[8] = {};
  float mrow[8], lrow[8];
  #pragma unroll
  for (int r = 0; r < 8; ++r) { mrow[r] = -1e30f; lrow[r] = 0.0f; }

  bf16_t* ps = &pshare[wid][0];
  const int ktmax_w   = (qt * 16 + 15) / 32;       // this wave's causal limit
  const int ktmax_blk = qblk * 4 + 3;              // block's last key tile

  stage(0, 0);
  int buf = 0;
  for (int kt = 0; kt <= ktmax_blk; ++kt) {
    async_join_barrier();                          // tiles in `buf` visible
    if (kt < ktmax_blk) stage(kt + 1, buf ^ 1);    // overlaps with compute

    if (kt <= ktmax_w) {                           // wave-uniform causal skip
      const bf16_t* shKb = &shK[buf][0];
      const bf16_t* shVb = &shV[buf][0];

      v8f s0 = {}, s1 = {};
      #pragma unroll
      for (int c = 0; c < 4; ++c) {
        BFrag kb0, kb1;
        const bf16_t* kc0 = shKb + idx * HD + c * 32;
        const bf16_t* kc1 = shKb + (16 + idx) * HD + c * 32;
        kb0.u[0] = *(const v4u*)(kc0 + 8 * half);
        kb0.u[1] = *(const v4u*)(kc0 + 16 + 8 * half);
        kb1.u[0] = *(const v4u*)(kc1 + 8 * half);
        kb1.u[1] = *(const v4u*)(kc1 + 16 + 8 * half);
        s0 = __builtin_amdgcn_wmma_f32_16x16x32_bf16(false, qf[c].v, false, kb0.v,
                                                     (short)0, s0, false, false);
        s1 = __builtin_amdgcn_wmma_f32_16x16x32_bf16(false, qf[c].v, false, kb1.v,
                                                     (short)0, s1, false, false);
      }

      const bool needmask = (kt * 32 + 31) > (qt * 16);
      float p0a[8], p1a[8], sf[8];
      #pragma unroll
      for (int r = 0; r < 8; ++r) {
        const int qg = qt * 16 + r + 8 * half;
        float e0 = s0[r] * sc;
        float e1 = s1[r] * sc;
        if (needmask) {
          if (kt * 32 + idx      > qg) e0 = -1e30f;
          if (kt * 32 + 16 + idx > qg) e1 = -1e30f;
        }
        float rm = fmaxf(e0, e1);
        rm = fmaxf(rm, __shfl_xor(rm, 1));   // masks < 16 stay within the half
        rm = fmaxf(rm, __shfl_xor(rm, 2));
        rm = fmaxf(rm, __shfl_xor(rm, 4));
        rm = fmaxf(rm, __shfl_xor(rm, 8));
        const float mn = fmaxf(mrow[r], rm);
        sf[r] = exp2f(mrow[r] - mn);
        mrow[r] = mn;
        p0a[r] = exp2f(e0 - mn);
        p1a[r] = exp2f(e1 - mn);
        lrow[r] = lrow[r] * sf[r] + p0a[r] + p1a[r];
      }
      #pragma unroll
      for (int n = 0; n < 8; ++n)
        #pragma unroll
        for (int r = 0; r < 8; ++r)
          o[n][r] *= sf[r];

      // P: C layout -> A layout round trip through this wave's LDS slice
      #pragma unroll
      for (int r = 0; r < 8; ++r) {
        ps[(r + 8 * half) * 32 + idx]      = (bf16_t)p0a[r];
        ps[(r + 8 * half) * 32 + 16 + idx] = (bf16_t)p1a[r];
      }
      BFrag pf;
      pf.u[0] = *(const v4u*)(ps + idx * 32 + 8 * half);
      pf.u[1] = *(const v4u*)(ps + idx * 32 + 16 + 8 * half);

      #pragma unroll
      for (int n = 0; n < 8; ++n) {
        const bf16_t* vr = shVb + (n * 16 + idx) * 32;
        BFrag vb;
        vb.u[0] = *(const v4u*)(vr + 8 * half);
        vb.u[1] = *(const v4u*)(vr + 16 + 8 * half);
        o[n] = __builtin_amdgcn_wmma_f32_16x16x32_bf16(false, pf.v, false, vb.v,
                                                       (short)0, o[n], false, false);
      }
    }
    buf ^= 1;
  }

  float rinv[8];
  #pragma unroll
  for (int r = 0; r < 8; ++r) {
    float L = lrow[r];
    L += __shfl_xor(L, 1);
    L += __shfl_xor(L, 2);
    L += __shfl_xor(L, 4);
    L += __shfl_xor(L, 8);
    rinv[r] = 1.0f / L;
  }
  #pragma unroll
  for (int n = 0; n < 8; ++n) {
    #pragma unroll
    for (int r = 0; r < 8; ++r) {
      const int trow = qt * 16 + r + 8 * half;
      attnbf[((size_t)(b * TT + trow)) * DM + h * HD + n * 16 + idx] =
          (bf16_t)(o[n][r] * rinv[r]);
    }
  }
}

// ---------------------------------------------------------------------------
extern "C" void kernel_launch(void* const* d_in, const int* in_sizes, int n_in,
                              void* d_out, int out_size, void* d_ws, size_t ws_size,
                              hipStream_t stream) {
  (void)in_sizes; (void)n_in; (void)out_size; (void)ws_size;
  const float* x    = (const float*)d_in[0];
  const float* wq   = (const float*)d_in[1];
  const float* wk   = (const float*)d_in[2];
  const float* wv   = (const float*)d_in[3];
  const float* wo   = (const float*)d_in[4];
  const float* cosb = (const float*)d_in[5];
  const float* sinb = (const float*)d_in[6];
  float* outp = (float*)d_out;

  char* ws = (char*)d_ws;
  size_t off = 0;
  auto alloc = [&](size_t bytes) -> void* {
    off = (off + 255) & ~(size_t)255;
    void* p = ws + off;
    off += bytes;
    return p;
  };

  const size_t M = (size_t)BB * TT;                 // 4096 rows
  bf16_t* xbf  = (bf16_t*)alloc(M * DM * sizeof(bf16_t));
  bf16_t* wqT  = (bf16_t*)alloc((size_t)DM * DM * sizeof(bf16_t));
  bf16_t* wkT  = (bf16_t*)alloc((size_t)(NKV * HD) * DM * sizeof(bf16_t));
  bf16_t* wvT  = (bf16_t*)alloc((size_t)(NKV * HD) * DM * sizeof(bf16_t));
  bf16_t* woT  = (bf16_t*)alloc((size_t)DM * DM * sizeof(bf16_t));
  float*  qf   = (float*)alloc(M * DM * sizeof(float));
  float*  kf   = (float*)alloc(M * (NKV * HD) * sizeof(float));
  float*  vf   = (float*)alloc(M * (NKV * HD) * sizeof(float));
  bf16_t* qbf  = (bf16_t*)alloc(M * DM * sizeof(bf16_t));
  bf16_t* kbf  = (bf16_t*)alloc(M * (NKV * HD) * sizeof(bf16_t));
  bf16_t* vtbf = (bf16_t*)alloc(M * (NKV * HD) * sizeof(bf16_t));
  bf16_t* attnbf = (bf16_t*)qf;   // qf is dead after RoPE; reuse for attn out

  cast_bf16_kernel<<<4096, 256, 0, stream>>>(x, xbf, M * DM);
  transpose_cast_kernel<<<dim3(DM / 32, DM / 32), 256, 0, stream>>>(wq, wqT, DM, DM);
  transpose_cast_kernel<<<dim3(DM / 32, (NKV * HD) / 32), 256, 0, stream>>>(wk, wkT, DM, NKV * HD);
  transpose_cast_kernel<<<dim3(DM / 32, (NKV * HD) / 32), 256, 0, stream>>>(wv, wvT, DM, NKV * HD);
  transpose_cast_kernel<<<dim3(DM / 32, DM / 32), 256, 0, stream>>>(wo, woT, DM, DM);

  gemm_bf16_wmma<<<(M / 256) * (DM / 64), 256, 0, stream>>>(xbf, wqT, qf, (int)M, DM, DM);
  gemm_bf16_wmma<<<(M / 256) * ((NKV * HD) / 64), 256, 0, stream>>>(xbf, wkT, kf, (int)M, NKV * HD, DM);
  gemm_bf16_wmma<<<(M / 256) * ((NKV * HD) / 64), 256, 0, stream>>>(xbf, wvT, vf, (int)M, NKV * HD, DM);

  rope_qk_kernel<<<(BB * TT * NH * HALF_D) / 256, 256, 0, stream>>>(qf, cosb, sinb, qbf, NH);
  rope_qk_kernel<<<(BB * TT * NKV * HALF_D) / 256, 256, 0, stream>>>(kf, cosb, sinb, kbf, NKV);
  vtrans_kernel<<<dim3(TT / 32, HD / 32, BB * NKV), 256, 0, stream>>>(vf, vtbf);

  flash_attn_kernel<<<(size_t)BB * NH * (TT / 128), 256, 0, stream>>>(qbf, kbf, vtbf, attnbf);

  gemm_bf16_wmma<<<(M / 256) * (DM / 64), 256, 0, stream>>>(attnbf, woT, outp, (int)M, DM, DM);
}
